// NerfRenderer_41188736368830
// MI455X (gfx1250) — matrix operand order repaired and checked by reading, not compiled, not run on hardware
//
#include <hip/hip_runtime.h>

typedef __attribute__((ext_vector_type(16))) _Float16 v16h;
typedef __attribute__((ext_vector_type(8)))  _Float16 v8h;
typedef __attribute__((ext_vector_type(2)))  _Float16 v2h;
typedef __attribute__((ext_vector_type(8)))  float    v8f;

#define NSAMP   128      // padded sample count (127 real)
#define HID     64
#define FD      32
#define GSZ     128
#define NEAR_T  0.1f
#define ETERM   1.0e-4f
#define LOG_257_256 0.0038986404f

#define NTILE_W2  4      // W2: kt{0,1} x nt{0,1}
#define NTILE_WR1 8      // Wr1: kt{0,1} x nt{0..3}
#define NTILES    (NTILE_W2 + NTILE_WR1)
#define TILE_HALFS 512   // 32 lanes x 16 halves per B tile

// ---- helpers -------------------------------------------------------------

// t-schedule: linspace(0.1, 1.1, 64) then 1.1 * (257/256)^(i-64)
__device__ __forceinline__ float tval(int i) {
    if (i < 64) return NEAR_T + (float)i * (1.0f / 63.0f);
    return 1.1f * __expf((float)(i - 64) * LOG_257_256);
}

__device__ __forceinline__ float softplusf(float x) {
    return (x > 20.0f) ? x : log1pf(__expf(x));
}
__device__ __forceinline__ float sigmoidf(float x) {
    return 1.0f / (1.0f + __expf(-x));
}

// Load a 16x32 f16 A-tile (rows = samples S0..S0+15, K = kb..kb+31) from LDS
// laid out row-major with row stride `stride` halves.
// Contiguous k order (same as packed B): lanes 0-15 hold K {kb+0..15}; lanes 16-31 hold
// K {kb+16..31}, for row M = lane&15.
__device__ __forceinline__ v16h load_a_tile(const _Float16* lds, int S0, int kb, int stride) {
    const int lane   = threadIdx.x & 31;
    const int row    = S0 + (lane & 15);
    const int firstK = kb + ((lane >> 4) ? 16 : 0);
    const _Float16* p = lds + row * stride + firstK;
    v8h lo = *(const v8h*)(p);
    v8h hi = *(const v8h*)(p + 8);
    v16h a;
#pragma unroll
    for (int i = 0; i < 8; ++i) { a[i] = lo[i]; a[8 + i] = hi[i]; }
    return a;
}

// Load a pre-packed f16 B tile (lane-major: tile*512 + lane*16 halves)
// with two global_load_b128 per lane.
__device__ __forceinline__ v16h load_b_packed(const _Float16* __restrict__ wp, int tile) {
    const int lane = threadIdx.x & 31;
    const v8h* p = (const v8h*)(wp + tile * TILE_HALFS + lane * 16);
    const v8h lo = p[0];
    const v8h hi = p[1];
    v16h b;
#pragma unroll
    for (int i = 0; i < 8; ++i) { b[i] = lo[i]; b[8 + i] = hi[i]; }
    return b;
}

#define WMMA_F16(A, B, C) \
    __builtin_amdgcn_wmma_f32_16x16x32_f16(false, (A), false, (B), (short)0, (C), false, false)

// ---- kernel 0: pack W2 / Wr1 into WMMA B-operand layout (f16) ------------
// Tile t, lane l, elem i maps to W[k][n] with n = nb + (l&15),
// k = kb + ((l>=16)?16:0) + i ; k >= kmax packs as zero.

__global__ __launch_bounds__(128) void pack_weights_kernel(
    const float* __restrict__ W2, const float* __restrict__ Wr1,
    _Float16* __restrict__ wp)
{
    for (int idx = threadIdx.x; idx < NTILES * 32; idx += 128) {
        const int tile = idx >> 5;
        const int lane = idx & 31;
        const float* W; int ldw, kb, nb, kmax;
        if (tile < NTILE_W2) {
            W = W2; ldw = FD; kmax = HID;
            kb = (tile >> 1) * 32; nb = (tile & 1) * 16;
        } else {
            const int t = tile - NTILE_W2;
            W = Wr1; ldw = HID; kmax = FD + 3;
            kb = (t >> 2) * 32; nb = (t & 3) * 16;
        }
        const int n  = nb + (lane & 15);
        const int k0 = kb + ((lane >> 4) ? 16 : 0);
        _Float16* dst = wp + tile * TILE_HALFS + lane * 16;
#pragma unroll
        for (int i = 0; i < 16; ++i) {
            const int k = k0 + i;
            dst[i] = (_Float16)((k < kmax) ? W[k * ldw + n] : 0.0f);
        }
    }
}

// ---- fused NeRF renderer: one block per ray ------------------------------

__global__ __launch_bounds__(128) void nerf_render_kernel(
    const float* __restrict__ rays_o, const float* __restrict__ rays_d,
    const float* __restrict__ grid,
    const float* __restrict__ W1,  const float* __restrict__ b1,
    const float* __restrict__ b2,
    const float* __restrict__ Ws,  const float* __restrict__ bs,
    const float* __restrict__ br1,
    const float* __restrict__ Wr2, const float* __restrict__ br2,
    const _Float16* __restrict__ wp,
    float* __restrict__ out)
{
    __shared__ _Float16 sh_h1[NSAMP * HID];   // layer1 activations   (16 KB)
    __shared__ _Float16 sh_fa[NSAMP * HID];   // feat(32)+dir(3)+pad  (16 KB)
    __shared__ _Float16 sh_h2[NSAMP * HID];   // rgb hidden           (16 KB)
    __shared__ float sh_w1[3 * HID];
    __shared__ float sh_b1[HID];
    __shared__ float sh_b2[FD];
    __shared__ float sh_ws[FD];
    __shared__ float sh_br1[HID];
    __shared__ float sh_wr2[HID * 3];
    __shared__ float sh_mask[NSAMP];
    __shared__ float sh_scan[NSAMP];
    __shared__ float sh_r[NSAMP];
    __shared__ float sh_g[NSAMP];
    __shared__ float sh_b[NSAMP];

    const int ray  = blockIdx.x;
    const int tid  = threadIdx.x;
    const int lane = tid & 31;
    const int wave = tid >> 5;
    const int s    = tid;                   // this thread's sample
    const bool valid = (s < 127);

    // ---- preload small weights into LDS ---------------------------------
    for (int i = tid; i < 3 * HID; i += 128) sh_w1[i] = W1[i];
    for (int i = tid; i < HID * 3; i += 128) sh_wr2[i] = Wr2[i];
    if (tid < HID) { sh_b1[tid] = b1[tid]; sh_br1[tid] = br1[tid]; }
    if (tid < FD)  { sh_b2[tid] = b2[tid]; sh_ws[tid]  = Ws[tid]; }

    // ---- per-sample geometry: position, contract, trilinear occupancy ---
    const float ox = rays_o[ray * 3 + 0], oy = rays_o[ray * 3 + 1], oz = rays_o[ray * 3 + 2];
    const float dx = rays_d[ray * 3 + 0], dy = rays_d[ray * 3 + 1], dz = rays_d[ray * 3 + 2];

    const float t    = tval(s);
    const float dist = tval(s + 1) - t;

    const float px = fmaf(dx, t, ox), py = fmaf(dy, t, oy), pz = fmaf(dz, t, oz);

    // mip360 contract, then /2
    const float nrm = fmaxf(fabsf(px), fmaxf(fabsf(py), fabsf(pz)));
    float cx = px, cy = py, cz = pz;
    if (nrm > 1.0f) {
        const float sfac = (2.0f - 1.0f / nrm) / nrm;
        cx *= sfac; cy *= sfac; cz *= sfac;
    }
    cx *= 0.5f; cy *= 0.5f; cz *= 0.5f;

    // trilinear sample of 128^3 occupancy grid
    const float gx = ((cx + 1.0f) * (float)GSZ - 1.0f) * 0.5f;
    const float gy = ((cy + 1.0f) * (float)GSZ - 1.0f) * 0.5f;
    const float gz = ((cz + 1.0f) * (float)GSZ - 1.0f) * 0.5f;
    const float fx0 = floorf(gx), fy0 = floorf(gy), fz0 = floorf(gz);
    const int x0 = (int)fx0, y0 = (int)fy0, z0 = (int)fz0;
    const float fx = gx - fx0, fy = gy - fy0, fz = gz - fz0;

    auto corner = [&](int zi, int yi, int xi) -> float {
        const bool ok = (zi >= 0) & (zi < GSZ) & (yi >= 0) & (yi < GSZ) & (xi >= 0) & (xi < GSZ);
        const int zc = min(max(zi, 0), GSZ - 1);
        const int yc = min(max(yi, 0), GSZ - 1);
        const int xc = min(max(xi, 0), GSZ - 1);
        const float v = grid[(zc * GSZ + yc) * GSZ + xc];
        return ok ? v : 0.0f;
    };
    const float occ =
        corner(z0,     y0,     x0    ) * (1.f-fz) * (1.f-fy) * (1.f-fx) +
        corner(z0,     y0,     x0 + 1) * (1.f-fz) * (1.f-fy) * fx +
        corner(z0,     y0 + 1, x0    ) * (1.f-fz) * fy       * (1.f-fx) +
        corner(z0,     y0 + 1, x0 + 1) * (1.f-fz) * fy       * fx +
        corner(z0 + 1, y0,     x0    ) * fz       * (1.f-fy) * (1.f-fx) +
        corner(z0 + 1, y0,     x0 + 1) * fz       * (1.f-fy) * fx +
        corner(z0 + 1, y0 + 1, x0    ) * fz       * fy       * (1.f-fx) +
        corner(z0 + 1, y0 + 1, x0 + 1) * fz       * fy       * fx;

    const float maskf = (valid && occ > 0.0f) ? 1.0f : 0.0f;
    sh_mask[s] = maskf;

    __syncthreads();   // sh_w1 / sh_b1 ready

    // ---- layer 1 (3 -> 64) on VALU, store packed f16 pairs to LDS -------
#pragma unroll 8
    for (int j = 0; j < HID; j += 2) {
        const float h0 = fmaf(cx, sh_w1[j],
                         fmaf(cy, sh_w1[HID + j],
                         fmaf(cz, sh_w1[2 * HID + j], sh_b1[j])));
        const float h1 = fmaf(cx, sh_w1[j + 1],
                         fmaf(cy, sh_w1[HID + j + 1],
                         fmaf(cz, sh_w1[2 * HID + j + 1], sh_b1[j + 1])));
        v2h pk;
        pk[0] = (_Float16)fmaxf(h0, 0.0f);
        pk[1] = (_Float16)fmaxf(h1, 0.0f);
        *(v2h*)&sh_h1[s * HID + j] = pk;
    }
    __syncthreads();

    // ---- layer 2 (64 -> 32) via WMMA: each wave owns 2 M-tiles ----------
    v16h B2[2][2];
#pragma unroll
    for (int kt = 0; kt < 2; ++kt)
#pragma unroll
        for (int nt = 0; nt < 2; ++nt)
            B2[kt][nt] = load_b_packed(wp, kt * 2 + nt);

#pragma unroll
    for (int mt = 0; mt < 2; ++mt) {
        const int S0 = wave * 32 + mt * 16;
        const v16h a0 = load_a_tile(sh_h1, S0, 0, HID);
        const v16h a1 = load_a_tile(sh_h1, S0, 32, HID);
#pragma unroll
        for (int nt = 0; nt < 2; ++nt) {
            v8f c = {};
            c = WMMA_F16(a0, B2[0][nt], c);
            c = WMMA_F16(a1, B2[1][nt], c);
            const int n    = nt * 16 + (lane & 15);
            const float bn = sh_b2[n];
            const int roff = ((lane >> 4) ? 8 : 0);
#pragma unroll
            for (int g = 0; g < 8; ++g) {
                const int row = S0 + g + roff;
                const float f = (c[g] + bn) * sh_mask[row];
                sh_fa[row * HID + n] = (_Float16)f;
            }
        }
    }
    // append dirs + zero pad (own row only; same wave as writer above)
    {
        v2h d01; d01[0] = (_Float16)dx; d01[1] = (_Float16)dy;
        v2h d23; d23[0] = (_Float16)dz; d23[1] = (_Float16)0.0f;
        v2h z2 = {};
        v8h z8 = {};
        _Float16* row = &sh_fa[s * HID];
        *(v2h*)(row + 32) = d01;
        *(v2h*)(row + 34) = d23;
        *(v2h*)(row + 36) = z2;
        *(v2h*)(row + 38) = z2;
        *(v8h*)(row + 40) = z8;
        *(v8h*)(row + 48) = z8;
        *(v8h*)(row + 56) = z8;
    }
    __syncthreads();

    // ---- sigma = softplus(feat @ Ws + bs) * mask ------------------------
    float sacc = bs[0];
    {
        const v8h* fp = (const v8h*)&sh_fa[s * HID];
#pragma unroll
        for (int ch = 0; ch < 4; ++ch) {
            const v8h v = fp[ch];
#pragma unroll
            for (int i = 0; i < 8; ++i)
                sacc = fmaf((float)v[i], sh_ws[ch * 8 + i], sacc);
        }
    }
    const float sigma = softplusf(sacc) * sh_mask[s];
    float alog = -sigma * dist;
    if (!valid) alog = 0.0f;

    // ---- transmittance: Hillis-Steele inclusive scan (log depth) --------
    sh_scan[s] = alog;
    __syncthreads();
#pragma unroll
    for (int off = 1; off < NSAMP; off <<= 1) {
        const float prev = (s >= off) ? sh_scan[s - off] : 0.0f;
        __syncthreads();
        sh_scan[s] += prev;
        __syncthreads();
    }
    const float csum  = (s > 0) ? sh_scan[s - 1] : 0.0f;   // exclusive sum
    const float trans = __expf(csum);
    const float alpha = 1.0f - __expf(alog);
    const float w     = trans * alpha;
    const float m2    = (sh_mask[s] > 0.0f && w > ETERM) ? 1.0f : 0.0f;
    const float wgt   = w * m2;

    // ---- rgb hidden layer ([feat|dir|0] 64 -> 64) via WMMA --------------
    v16h BR[2][4];
#pragma unroll
    for (int kt = 0; kt < 2; ++kt)
#pragma unroll
        for (int nt = 0; nt < 4; ++nt)
            BR[kt][nt] = load_b_packed(wp, NTILE_W2 + kt * 4 + nt);

#pragma unroll
    for (int mt = 0; mt < 2; ++mt) {
        const int S0 = wave * 32 + mt * 16;
        const v16h a0 = load_a_tile(sh_fa, S0, 0, HID);
        const v16h a1 = load_a_tile(sh_fa, S0, 32, HID);
#pragma unroll
        for (int nt = 0; nt < 4; ++nt) {
            v8f c = {};
            c = WMMA_F16(a0, BR[0][nt], c);
            c = WMMA_F16(a1, BR[1][nt], c);
            const int n    = nt * 16 + (lane & 15);
            const float bn = sh_br1[n];
            const int roff = ((lane >> 4) ? 8 : 0);
#pragma unroll
            for (int g = 0; g < 8; ++g) {
                const int row = S0 + g + roff;
                sh_h2[row * HID + n] = (_Float16)fmaxf(c[g] + bn, 0.0f);
            }
        }
    }
    __syncthreads();

    // ---- rgb head (64 -> 3), sigmoid, weight ----------------------------
    float rr = br2[0], gg = br2[1], bb = br2[2];
    {
        const v8h* hp = (const v8h*)&sh_h2[s * HID];
#pragma unroll
        for (int ch = 0; ch < 8; ++ch) {
            const v8h v = hp[ch];
#pragma unroll
            for (int i = 0; i < 8; ++i) {
                const float hv = (float)v[i];
                const int k = ch * 8 + i;
                rr = fmaf(hv, sh_wr2[k * 3 + 0], rr);
                gg = fmaf(hv, sh_wr2[k * 3 + 1], gg);
                bb = fmaf(hv, sh_wr2[k * 3 + 2], bb);
            }
        }
    }
    rr = sigmoidf(rr) * wgt;
    gg = sigmoidf(gg) * wgt;
    bb = sigmoidf(bb) * wgt;

    // ---- deterministic tree reduction over samples ----------------------
    sh_r[s] = rr; sh_g[s] = gg; sh_b[s] = bb;
    __syncthreads();
#pragma unroll
    for (int st = 64; st > 0; st >>= 1) {
        if (tid < st) {
            sh_r[tid] += sh_r[tid + st];
            sh_g[tid] += sh_g[tid + st];
            sh_b[tid] += sh_b[tid + st];
        }
        __syncthreads();
    }
    if (tid == 0) {
        out[ray * 3 + 0] = sh_r[0];
        out[ray * 3 + 1] = sh_g[0];
        out[ray * 3 + 2] = sh_b[0];
    }
}

extern "C" void kernel_launch(void* const* d_in, const int* in_sizes, int n_in,
                              void* d_out, int out_size, void* d_ws, size_t ws_size,
                              hipStream_t stream) {
    (void)n_in; (void)ws_size; (void)out_size;
    const float* rays_o = (const float*)d_in[0];
    const float* rays_d = (const float*)d_in[1];
    const float* grid   = (const float*)d_in[2];
    const float* W1     = (const float*)d_in[3];
    const float* b1     = (const float*)d_in[4];
    const float* W2     = (const float*)d_in[5];
    const float* b2     = (const float*)d_in[6];
    const float* Ws     = (const float*)d_in[7];
    const float* bs     = (const float*)d_in[8];
    const float* Wr1    = (const float*)d_in[9];
    const float* br1    = (const float*)d_in[10];
    const float* Wr2    = (const float*)d_in[11];
    const float* br2    = (const float*)d_in[12];
    float* out = (float*)d_out;
    _Float16* wpack = (_Float16*)d_ws;       // 12 KB of packed f16 B tiles

    pack_weights_kernel<<<1, 128, 0, stream>>>(W2, Wr1, wpack);

    const int n_rays = in_sizes[0] / 3;
    nerf_render_kernel<<<n_rays, 128, 0, stream>>>(
        rays_o, rays_d, grid, W1, b1, b2, Ws, bs, br1, Wr2, br2, wpack, out);
}
